// MLPEdgeDecoder_70849780515149
// MI455X (gfx1250) — matrix operand order repaired and checked by reading, not compiled, not run on hardware
//
#include <hip/hip_runtime.h>
#include <hip/hip_bf16.h>

typedef __attribute__((ext_vector_type(16))) _Float16 v16h;
typedef __attribute__((ext_vector_type(8)))  float    v8f;

#define HIDDEN 128     // N dimension and per-node feature width
#define NKSTEP 8       // K = 256 = 8 * 32
#define NNTILE 8       // N = 128 = 8 * 16
#define MTILES 4       // 4 x 16-pair M tiles per wave (B-fragment reuse x4)

// ---------------------------------------------------------------------------
// One wave handles 64 pairs (four 16-row M tiles). Block = 8 waves = 512 pairs.
// LDS holds W1 (256x128) converted to f16, pre-swizzled into the WMMA
// B-fragment layout so the inner loop reads it with plain ds_load_b128.
// Each B fragment is reused for all four M tiles -> 256 B of LDS traffic per
// WMMA, ~128 B/clk per WGP at full matrix-pipe rate (half the 256 B/clk LDS
// ceiling).
//
// B-fragment layout (16-bit B, 32x16): element (k, n) of a 32-wide K chunk:
//   lane = (n % 16) + 16*((k / 8) % 2), slot within lane's 16 halves = (k % 8) + 8*(k / 16).
// A-fragment layout (16-bit A, 16x32): lane = m + 16*hf; lane holds
//   K = ks*32 + hf*8 + [0..8) in VGPR0-3 and K = ks*32 + 16 + hf*8 + [0..8)
//   in VGPR4-7  -> two contiguous 32-byte runs per lane (float4 friendly).
// ---------------------------------------------------------------------------
__global__ __launch_bounds__(256) void edge_mlp_kernel(
    const float* __restrict__ z,
    const long long* __restrict__ pairs,   // int64 (NUM_PAIRS, 2)
    const float* __restrict__ W1,          // (256, 128) row-major
    const float* __restrict__ b1,          // (128,)
    const float* __restrict__ W2,          // (128, 1)
    const float* __restrict__ b2,          // (1,)
    float* __restrict__ out,               // (NUM_PAIRS,)
    int n_pairs, int n_tiles)              // n_tiles = ceil(n_pairs / 64)
{
    __shared__ __align__(32) _Float16 sW1[256 * 128];   // 64 KB

    const int t = threadIdx.x;

    // ---- Stage W1 -> LDS (f32 -> f16, swizzled into B-fragment order) ----
    // Coalesced: 256 threads * float4 = 4 KB contiguous per iteration.
    for (int s = 0; s < 32; ++s) {
        const int i4 = (s * 256 + t) * 4;          // linear f32 index, n0 % 4 == 0
        const int k  = i4 >> 7;                    // 0..255
        const int n0 = i4 & 127;
        const float4 w = *(const float4*)(W1 + i4);
        const int ks  = k >> 5;                    // K chunk 0..7
        const int kl  = k & 31;
        const int e   = (kl & 7) | ((kl >> 4) << 3); // slot within lane
        const int khi = (kl >> 3) & 1;             // which 16-lane half
        const float wv[4] = {w.x, w.y, w.z, w.w};
        #pragma unroll
        for (int q = 0; q < 4; ++q) {
            const int n    = n0 + q;
            const int nt   = n >> 4;
            const int l    = (n & 15) + (khi << 4);
            const int cell = ((ks << 3) + nt) * 32 + l;   // fragment cell
            sW1[cell * 16 + e] = (_Float16)wv[q];
        }
    }
    __syncthreads();

    const int wave = t >> 5;
    const int lane = t & 31;
    const int tile = blockIdx.x * 8 + wave;
    if (tile >= n_tiles) return;                   // wave-uniform; no more barriers

    const int m  = lane & 15;                      // row (pair) within an M tile
    const int hf = lane >> 4;                      // A-layout half select
    const int pbase = tile * (16 * MTILES);

    const float* rowu[MTILES];
    const float* rowv[MTILES];
    #pragma unroll
    for (int mt = 0; mt < MTILES; ++mt) {
        int p = pbase + mt * 16 + m;
        if (p >= n_pairs) p = n_pairs - 1;         // clamp: keep EXEC all-ones
        const long long nu = pairs[2 * (long long)p];
        const long long nv = pairs[2 * (long long)p + 1];
        rowu[mt] = z + (size_t)nu * HIDDEN;
        rowv[mt] = z + (size_t)nv * HIDDEN;
    }

    v8f acc[MTILES][NNTILE];
    const v8f vzero = {0.f, 0.f, 0.f, 0.f, 0.f, 0.f, 0.f, 0.f};
    #pragma unroll
    for (int mt = 0; mt < MTILES; ++mt)
        #pragma unroll
        for (int i = 0; i < NNTILE; ++i) acc[mt][i] = vzero;

    #pragma unroll
    for (int ks = 0; ks < NKSTEP; ++ks) {
        // Gather A fragments: two contiguous 32B runs from z_u (ks<4) or z_v.
        v16h a[MTILES];
        #pragma unroll
        for (int mt = 0; mt < MTILES; ++mt) {
            const float* rp   = (ks < 4) ? rowu[mt] : rowv[mt];
            const float* base = rp + (ks & 3) * 32 + hf * 8;
            const float4 f0 = *(const float4*)(base);
            const float4 f1 = *(const float4*)(base + 4);
            const float4 f2 = *(const float4*)(base + 16);
            const float4 f3 = *(const float4*)(base + 20);
            a[mt][0]  = (_Float16)f0.x; a[mt][1]  = (_Float16)f0.y;
            a[mt][2]  = (_Float16)f0.z; a[mt][3]  = (_Float16)f0.w;
            a[mt][4]  = (_Float16)f1.x; a[mt][5]  = (_Float16)f1.y;
            a[mt][6]  = (_Float16)f1.z; a[mt][7]  = (_Float16)f1.w;
            a[mt][8]  = (_Float16)f2.x; a[mt][9]  = (_Float16)f2.y;
            a[mt][10] = (_Float16)f2.z; a[mt][11] = (_Float16)f2.w;
            a[mt][12] = (_Float16)f3.x; a[mt][13] = (_Float16)f3.y;
            a[mt][14] = (_Float16)f3.z; a[mt][15] = (_Float16)f3.w;
        }

        #pragma unroll
        for (int nt = 0; nt < NNTILE; ++nt) {
            const v16h bfrag =
                *(const v16h*)&sW1[(((ks << 3) + nt) * 32 + lane) * 16];
            #pragma unroll
            for (int mt = 0; mt < MTILES; ++mt)
                acc[mt][nt] = __builtin_amdgcn_wmma_f32_16x16x32_f16(
                    false, a[mt], false, bfrag, (short)0, acc[mt][nt],
                    false, false);
        }
    }

    // ---- Epilogue: bias + ReLU + dot with W2, fused per C fragment ----
    // C layout: lane holds column n = nt*16 + col for rows hi*8 + j (j = VGPR).
    const int col = lane & 15;
    const int hi  = lane >> 4;
    #pragma unroll
    for (int mt = 0; mt < MTILES; ++mt) {
        float part[8] = {0.f, 0.f, 0.f, 0.f, 0.f, 0.f, 0.f, 0.f};
        #pragma unroll
        for (int nt = 0; nt < NNTILE; ++nt) {
            const int n = (nt << 4) + col;
            const float b1n = b1[n];
            const float w2n = W2[n];
            #pragma unroll
            for (int j = 0; j < 8; ++j) {
                float h = acc[mt][nt][j] + b1n;
                h = h > 0.f ? h : 0.f;
                part[j] += h * w2n;
            }
        }
        // Reduce across the 16 lanes of each half (sums over n % 16).
        #pragma unroll
        for (int mask = 1; mask < 16; mask <<= 1) {
            #pragma unroll
            for (int j = 0; j < 8; ++j)
                part[j] += __shfl_xor(part[j], mask, 32);
        }
        // Lane (hi*16 + j), j in [0,8), writes row hi*8 + j of this M tile.
        float r = part[0];
        r = (col == 1) ? part[1] : r;
        r = (col == 2) ? part[2] : r;
        r = (col == 3) ? part[3] : r;
        r = (col == 4) ? part[4] : r;
        r = (col == 5) ? part[5] : r;
        r = (col == 6) ? part[6] : r;
        r = (col == 7) ? part[7] : r;
        if (col < 8) {
            const int prow = pbase + mt * 16 + hi * 8 + col;
            if (prow < n_pairs) out[prow] = r + b2[0];
        }
    }
}

extern "C" void kernel_launch(void* const* d_in, const int* in_sizes, int n_in,
                              void* d_out, int out_size, void* d_ws, size_t ws_size,
                              hipStream_t stream) {
    const float*     z     = (const float*)d_in[0];
    const long long* pairs = (const long long*)d_in[1];   // int64 (P,2)
    const float*     W1    = (const float*)d_in[2];
    const float*     b1    = (const float*)d_in[3];
    const float*     W2    = (const float*)d_in[4];
    const float*     b2    = (const float*)d_in[5];
    float*           out   = (float*)d_out;

    const int n_pairs = in_sizes[1] / 2;
    const int n_tiles = (n_pairs + (16 * MTILES) - 1) / (16 * MTILES);
    const int blocks  = (n_tiles + 7) / 8;

    edge_mlp_kernel<<<blocks, 256, 0, stream>>>(z, pairs, W1, b1, W2, b2,
                                                out, n_pairs, n_tiles);
}
